// DecoderRNN_61813169324368
// MI455X (gfx1250) — compile-verified, matrix operand order checked by reading, not run on hardware
//
#include <hip/hip_runtime.h>
#include <hip/hip_bf16.h>
#include <math.h>

typedef __attribute__((ext_vector_type(16))) _Float16 v16h;
typedef __attribute__((ext_vector_type(8)))  _Float16 v8h;
typedef __attribute__((ext_vector_type(8)))  float    v8f;

#define KC 256                    // K-chunk staged in LDS
#define LDS_STRIDE (KC + 8)       // +8 halves pad: 528B row stride -> conflict-free

// ---------------------------------------------------------------------------
// Pack a KxN row-major f32 weight matrix into WMMA B-fragment order (f16).
// Layout: [kTile][nTile][lane][16 halves], 32B contiguous per lane per frag.
// B-fragment (16-bit, 32x16 KxN), per CDNA5 ISA 7.12.2:
//   lane = n%16 + 16*bit3(k);  half = (k&7) + 8*bit4(k)
// ---------------------------------------------------------------------------
__global__ void pack_b_f16(const float* __restrict__ W, _Float16* __restrict__ out,
                           int K, int N) {
  long long gid = (long long)blockIdx.x * blockDim.x + threadIdx.x;
  long long total = (long long)K * N;
  if (gid >= total) return;
  int k = (int)(gid / N);
  int n = (int)(gid - (long long)(gid / N) * N);
  int kTile = k >> 5, kIn = k & 31;
  int nTile = n >> 4, n16 = n & 15;
  int g    = (kIn >> 3) & 1;
  int half = (kIn & 7) + 8 * ((kIn >> 4) & 1);
  int lane = n16 + 16 * g;
  int nTiles = N >> 4;
  long long dst = ((long long)(kTile * nTiles + nTile) * 32 + lane) * 16 + half;
  out[dst] = (_Float16)W[(long long)k * N + n];
}

// ---------------------------------------------------------------------------
// Build X f16 (rows = b*T + t, cols = E): t==0 -> image features, else embed.
// ---------------------------------------------------------------------------
__global__ void gather_x_f16(const float* __restrict__ features,
                             const int* __restrict__ captions,
                             const float* __restrict__ embed,
                             _Float16* __restrict__ X,
                             int B, int T, int E, int Tm1) {
  long long gid = (long long)blockIdx.x * blockDim.x + threadIdx.x;
  long long total = (long long)B * T * E;
  if (gid >= total) return;
  int e   = (int)(gid % E);
  int row = (int)(gid / E);
  int b = row / T;
  int t = row - b * T;
  float v;
  if (t == 0) {
    v = features[(long long)b * E + e];
  } else {
    int idx = captions[b * Tm1 + (t - 1)];
    v = embed[(long long)idx * E + e];
  }
  X[(long long)row * E + e] = (_Float16)v;
}

// ---------------------------------------------------------------------------
// WMMA GEMM with LDS-staged A: C(MxN) = A(MxK f16) * Bpacked(KxN) [+ epilogue]
// Block = 256 threads = 8 waves. All 8 waves share one 64-row M-group whose A
// tile is cooperatively staged into LDS in 256-wide K chunks; each wave owns
// one 16-wide N tile and computes 4 stacked 16x16 output tiles.
// Per k-step: one B fragment (incrementing-pointer global load + prefetch of
// the next), then ALL FOUR A fragments loaded from LDS into distinct
// registers before four back-to-back WMMAs (independent accumulators).
// A row index for output row m is m*aRowMul + aRowAdd (lets the recurrence
// read hs row b*32+(t-1) while producing row b*32+t).
// mode 0: outF[m*N+n] = acc + bias[n]
// mode 1: h = tanh(acc + xadd[orow*N+n] + bias[n]); outH[orow*N+n] = (f16)h,
//         orow = m*aRowMul + (aRowAdd+1)
// ---------------------------------------------------------------------------
__global__ void wmma_gemm_f16(const _Float16* __restrict__ A,
                              int aRowMul, int aRowAdd,
                              const _Float16* __restrict__ Bp,
                              const float* __restrict__ bias,
                              const float* __restrict__ xadd,
                              float* __restrict__ outF,
                              _Float16* __restrict__ outH,
                              int M, int N, int K, int mode, int skipA) {
  __shared__ _Float16 smem[64 * LDS_STRIDE];   // 33.8 KB

  int tid  = threadIdx.x;
  int lane = tid & 31;
  int wave = tid >> 5;

  int nTiles   = N >> 4;
  int mGroups  = M >> 6;
  int nBlocksN = (nTiles + 7) >> 3;
  int mGroup   = blockIdx.x / nBlocksN;
  int nTile    = (blockIdx.x - mGroup * nBlocksN) * 8 + wave;
  bool active  = (mGroup < mGroups) && (nTile < nTiles);  // wave-uniform

  int m16 = lane & 15;
  int g   = lane >> 4;

  v8f acc[4] = {v8f{}, v8f{}, v8f{}, v8f{}};

  if (!skipA) {
    // Packed-B walk: address advances by a constant nTiles*512 halves / kTile.
    const long long bStride = (long long)nTiles * 512;
    const _Float16* bptr =
        Bp + (long long)nTile * 512 + (long long)lane * 16;

    for (int kc = 0; kc < K; kc += KC) {
      __syncthreads();
      // Cooperative stage: 64 rows x KC halves; 16B chunks, coalesced rows.
      for (int i = tid; i < 64 * (KC / 8); i += 256) {
        int row = i >> 5;                 // 32 chunks of 8 halves per row
        int c8  = i & 31;
        long long grow = (long long)(mGroup * 64 + row) * aRowMul + aRowAdd;
        v8h chunk = *(const v8h*)(A + grow * K + kc + c8 * 8);
        *(v8h*)(smem + row * LDS_STRIDE + c8 * 8) = chunk;
      }
      __syncthreads();

      if (active) {
        for (int kt = 0; kt < (KC >> 5); ++kt) {
          v16h bfrag = *(const v16h*)bptr;
          __builtin_prefetch((const void*)(bptr + bStride), 0, 3);
          int k0 = kt << 5;

          // A fragments from LDS (16-bit 16x32 MxK layout):
          // lanes 0-15: K {k0..k0+7, k0+16..k0+23}; lanes 16-31: +8.
          v16h af[4];
#pragma unroll
          for (int mi = 0; mi < 4; ++mi) {
            const _Float16* arow = smem + (mi * 16 + m16) * LDS_STRIDE + k0;
            v8h lo = *(const v8h*)(arow + 8 * g);
            v8h hi = *(const v8h*)(arow + 16 + 8 * g);
            af[mi] = __builtin_shufflevector(lo, hi, 0, 1, 2, 3, 4, 5, 6, 7,
                                             8, 9, 10, 11, 12, 13, 14, 15);
          }
#pragma unroll
          for (int mi = 0; mi < 4; ++mi) {
            acc[mi] = __builtin_amdgcn_wmma_f32_16x16x32_f16(
                false, af[mi], false, bfrag, (short)0, acc[mi], false, false);
          }
          bptr += bStride;
        }
      } else {
        bptr += bStride * (KC >> 5);   // keep pointer in sync (dead anyway)
      }
    }
  }

  if (!active) return;

  // Epilogue. D layout: lane col = lane&15, VGPR r holds row r + 8*(lane>>4).
  int nCol = nTile * 16 + m16;
  float bv = bias[nCol];
#pragma unroll
  for (int mi = 0; mi < 4; ++mi) {
#pragma unroll
    for (int r = 0; r < 8; ++r) {
      int row = mGroup * 64 + mi * 16 + r + 8 * g;
      float v = acc[mi][r] + bv;
      if (mode == 1) {
        long long orow = (long long)row * aRowMul + (aRowAdd + 1);
        v = tanhf(v + xadd[orow * (long long)N + nCol]);
        outH[orow * (long long)N + nCol] = (_Float16)v;
      } else {
        outF[(long long)row * N + nCol] = v;
      }
    }
  }
}

extern "C" void kernel_launch(void* const* d_in, const int* in_sizes, int n_in,
                              void* d_out, int out_size, void* d_ws, size_t ws_size,
                              hipStream_t stream) {
  (void)in_sizes; (void)n_in; (void)out_size; (void)ws_size;
  const int B = 64, T = 32, E = 512, H = 512, V = 10000, Tm1 = 31;

  const float* features = (const float*)d_in[0];
  const int*   captions = (const int*)d_in[1];
  const float* embed    = (const float*)d_in[2];
  const float* Wi       = (const float*)d_in[3];
  const float* bi       = (const float*)d_in[4];
  const float* Wh       = (const float*)d_in[5];
  const float* bh       = (const float*)d_in[6];
  const float* Wy       = (const float*)d_in[7];
  const float* by       = (const float*)d_in[8];
  float* out = (float*)d_out;

  // Workspace carve-up (256B aligned), ~20 MB total.
  char* ws = (char*)d_ws;
  size_t off = 0;
  auto carve = [&](size_t bytes) {
    char* p = ws + off;
    off = (off + bytes + 255) & ~(size_t)255;
    return p;
  };
  _Float16* WiP   = (_Float16*)carve((size_t)E * H * 2);     // 512 KB
  _Float16* WhP   = (_Float16*)carve((size_t)H * H * 2);     // 512 KB
  _Float16* WyP   = (_Float16*)carve((size_t)H * V * 2);     // 10.24 MB
  _Float16* X16   = (_Float16*)carve((size_t)B * T * E * 2); // 2 MB
  _Float16* hs16  = (_Float16*)carve((size_t)B * T * H * 2); // 2 MB
  float*    xproj = (float*)carve((size_t)B * T * H * 4);    // 4 MB

  const int TPB = 256;
  // 1) Pack weights into B-fragment order (f32 -> f16).
  pack_b_f16<<<(E * H + TPB - 1) / TPB, TPB, 0, stream>>>(Wi, WiP, E, H);
  pack_b_f16<<<(H * H + TPB - 1) / TPB, TPB, 0, stream>>>(Wh, WhP, H, H);
  pack_b_f16<<<((long long)H * V + TPB - 1) / TPB, TPB, 0, stream>>>(Wy, WyP, H, V);
  // 2) Gather/convert inputs: rows b*T+t.
  gather_x_f16<<<((long long)B * T * E + TPB - 1) / TPB, TPB, 0, stream>>>(
      features, captions, embed, X16, B, T, E, Tm1);

  auto blocksFor = [](int Mr, int Nr) { // (M/64) m-groups x ceil((N/16)/8)
    int mGroups  = Mr >> 6;
    int nBlocksN = ((Nr >> 4) + 7) >> 3;
    return mGroups * nBlocksN;
  };

  // 3) xproj = X @ Wi + bi          (2048x512x512)
  wmma_gemm_f16<<<blocksFor(B * T, H), TPB, 0, stream>>>(
      X16, 1, 0, WiP, bi, nullptr, xproj, nullptr, B * T, H, E, /*mode=*/0, /*skipA=*/0);

  // 4) Recurrence: hs[b*32+t] = tanh(xproj[b*32+t] + hs[b*32+t-1] @ Wh + bh)
  for (int t = 0; t < T; ++t) {
    wmma_gemm_f16<<<blocksFor(B, H), TPB, 0, stream>>>(
        hs16, /*aRowMul=*/T, /*aRowAdd=*/t - 1, WhP, bh, xproj,
        nullptr, hs16, B, H, H, /*mode=*/1, /*skipA=*/(t == 0) ? 1 : 0);
  }

  // 5) logits = hs @ Wy + by        (2048x10000x512) -> d_out (B,T,V)
  wmma_gemm_f16<<<blocksFor(B * T, V), TPB, 0, stream>>>(
      hs16, 1, 0, WyP, by, nullptr, out, nullptr, B * T, V, H, /*mode=*/0, /*skipA=*/0);
}
